// OrtonEffect_8315056685432
// MI455X (gfx1250) — matrix operand order repaired and checked by reading, not compiled, not run on hardware
//
#include <hip/hip_runtime.h>
#include <math.h>

typedef __attribute__((ext_vector_type(16))) _Float16 v16h;
typedef __attribute__((ext_vector_type(8)))  _Float16 v8h;
typedef __attribute__((ext_vector_type(8)))  float    v8f;
typedef __attribute__((ext_vector_type(4)))  float    v4f;

#define IMG_H 1024
#define IMG_W 1024
#define NCHAN 24             // 8 images * 3 channels
#define TAPS  51
#define SHIFT 7              // Toeplitz shift: conv window starts at -32 (8-aligned)
#define FRONT 32             // front pad of intermediate rows (multiple of 8)
#define HP    1104           // padded row: 32 front + 1024 data + 48 back (16B-aligned rows)
#define TX    64             // 16-wide tiles per row
#define TY    64             // 16-tall tiles per column
#define TPW   4              // tiles per wave
#define NWAVES (NCHAN * TY * (TX / TPW))   // 24576 waves per pass

// Normalized 1D gaussian in LDS (sigma=15, 51 taps, matches reference).
__device__ inline void build_weights(float* sg, float* snorm) {
    if (threadIdx.x < TAPS) {
        float d = (float)((int)threadIdx.x - 25);
        sg[threadIdx.x] = expf(-d * d * (1.0f / (2.0f * 15.0f * 15.0f)));
    }
    __syncthreads();
    if (threadIdx.x == 0) {
        float s = 0.0f;
        for (int i = 0; i < TAPS; ++i) s += sg[i];
        *snorm = 1.0f / s;
    }
    __syncthreads();
}

// Lane slice of the shifted 96x16 Toeplitz B[j][n] = g[j - n - SHIFT].
// CDNA5 16-bit B (32x16): lanes 0-15 hold K=0..7,16..23 of col N=lane,
// lanes 16-31 hold K=8..15,24..31 of col N=lane-16.
__device__ inline v16h toeplitz_chunk(const float* sg, float sn, int kc, int nn, int hs) {
    v16h b;
#pragma unroll
    for (int e = 0; e < 16; ++e) {
        int t = kc * 32 + 8 * hs + e + (e >= 8 ? 8 : 0) - nn - SHIFT;
        float w = (t >= 0 && t < TAPS) ? sg[t] * sn : 0.0f;
        b[e] = (_Float16)w;
    }
    return b;
}

// Zero the pad columns of the transposed intermediate (front 32, back 48).
__global__ void orton_zero_pads(_Float16* __restrict__ tmpT) {
    int tid = blockIdx.x * blockDim.x + threadIdx.x;
    const int total = NCHAN * IMG_W * 80;
    if (tid >= total) return;
    int row = tid / 80;
    int k   = tid - row * 80;
    int idx = (k < FRONT) ? k : (IMG_H + k);   // [0,32) and [1056,1104)
    tmpT[(size_t)row * HP + idx] = (_Float16)0.0f;
}

// Pass 1: horizontal 51-tap conv; result stored transposed as f16 at
// tmpT[x][FRONT + y] with a single aligned b128 store per lane per tile.
__global__ __launch_bounds__(256) void orton_pass1(const float* __restrict__ x,
                                                   _Float16* __restrict__ tmpT) {
    __shared__ float sg[TAPS];
    __shared__ float snorm;
    build_weights(sg, &snorm);

    const int wid = (int)((blockIdx.x * blockDim.x + threadIdx.x) >> 5);
    if (wid >= NWAVES) return;
    const int lane = (int)(threadIdx.x & 31);
    const int nn   = lane & 15;     // B/D column (x offset)
    const int hs   = lane >> 4;

    const int TXG = TX / TPW;       // 16 groups of 4 x-tiles
    const int img = wid / (TY * TXG);
    const int rem = wid - img * (TY * TXG);
    const int ty  = rem / TXG;
    const int txg = rem - ty * TXG;
    const int y0  = ty * 16;

    const float sn = snorm;
    v16h bm[3];
#pragma unroll
    for (int kc = 0; kc < 3; ++kc) bm[kc] = toeplitz_chunk(sg, sn, kc, nn, hs);

    // A row for this lane (M = lane&15) is fixed across the 4 tiles.
    const float* rp = x + (size_t)img * (IMG_H * IMG_W) + (size_t)(y0 + nn) * IMG_W;

    for (int i = 0; i < TPW; ++i) {
        const int x0 = (txg * TPW + i) * 16;
        // Issue ALL 12 run-loads for this tile up front (one clause, full MLP).
        // Run r6 = kc*2 + run starts at column x0 - 32 + 8*hs + 16*r6.
        v4f f[12];
        if (x0 >= 32 && x0 <= IMG_W - 64) {
            // Interior (uniform, 59/64): all runs in range, immediate offsets.
            const float* base = rp + (x0 - 32 + 8 * hs);
#pragma unroll
            for (int r6 = 0; r6 < 6; ++r6) {
                f[2 * r6]     = *(const v4f*)(base + 16 * r6);
                f[2 * r6 + 1] = *(const v4f*)(base + 16 * r6 + 4);
            }
        } else {
            // Border: 8-aligned runs are entirely in or out of the image.
#pragma unroll
            for (int r6 = 0; r6 < 6; ++r6) {
                const int c0 = x0 - 32 + 8 * hs + 16 * r6;
                const bool in = (c0 >= 0) && (c0 < IMG_W);
                const int cc = in ? c0 : 0;          // safe address, always load
                v4f f0 = *(const v4f*)(rp + cc);
                v4f f1 = *(const v4f*)(rp + cc + 4);
#pragma unroll
                for (int e = 0; e < 4; ++e) {
                    f[2 * r6][e]     = in ? f0[e] : 0.0f;
                    f[2 * r6 + 1][e] = in ? f1[e] : 0.0f;
                }
            }
        }
        v8f acc = {};
#pragma unroll
        for (int kc = 0; kc < 3; ++kc) {
            v16h a;
#pragma unroll
            for (int run = 0; run < 2; ++run) {
                const int r6 = kc * 2 + run;
#pragma unroll
                for (int e = 0; e < 4; ++e) {
                    a[run * 8 + e]     = (_Float16)f[2 * r6][e];
                    a[run * 8 + 4 + e] = (_Float16)f[2 * r6 + 1][e];
                }
            }
            acc = __builtin_amdgcn_wmma_f32_16x16x32_f16(false, a, false, bm[kc],
                                                         (short)0, acc, false, false);
        }
        v8h hv;
#pragma unroll
        for (int r = 0; r < 8; ++r) hv[r] = (_Float16)acc[r];
        _Float16* tp = tmpT + (size_t)img * (IMG_W * HP) + (size_t)(x0 + nn) * HP
                     + (FRONT + y0 + 8 * hs);        // 16B aligned
        *(v8h*)tp = hv;
    }
}

// Pass 2: vertical 51-tap conv over contiguous padded tmpT rows (guard-free
// aligned b128 loads), fused sigmoid-amount screen blend + clamp, NT stores.
__global__ __launch_bounds__(256) void orton_pass2(const float* __restrict__ x,
                                                   const _Float16* __restrict__ tmpT,
                                                   const float* __restrict__ amount_p,
                                                   float* __restrict__ out) {
    __shared__ float sg[TAPS];
    __shared__ float snorm;
    build_weights(sg, &snorm);

    const int wid = (int)((blockIdx.x * blockDim.x + threadIdx.x) >> 5);
    if (wid >= NWAVES) return;
    const int lane = (int)(threadIdx.x & 31);
    const int nn   = lane & 15;     // A row m (x offset) AND D column (y offset)
    const int hs   = lane >> 4;

    const int TYG = TY / TPW;       // 16 groups of 4 y-tiles
    const int img = wid / (TYG * TX);
    const int rem = wid - img * (TYG * TX);
    const int tyg = rem / TX;
    const int tx  = rem - tyg * TX;
    const int x0  = tx * 16;

    const float sn = snorm;
    v16h bm[3];
#pragma unroll
    for (int kc = 0; kc < 3; ++kc) bm[kc] = toeplitz_chunk(sg, sn, kc, nn, hs);

    const float amount = 0.4f / (1.0f + expf(-amount_p[0]));
    const float s = 1.2f * amount;

    // A row (tmpT row x0+m) is fixed across the 4 y-tiles -> strong locality.
    const _Float16* trow = tmpT + (size_t)img * (IMG_W * HP) + (size_t)(x0 + nn) * HP;

    for (int i = 0; i < TPW; ++i) {
        const int y0 = (tyg * TPW + i) * 16;

        // Issue ALL tile loads up front: 6 aligned b128 from tmpT (pads cover
        // the halo, no guards) + 2 b128 of x for the blend.
        const _Float16* tb = trow + (y0 + 8 * hs);   // multiple of 8 -> 16B aligned
        v8h rr[6];
#pragma unroll
        for (int r6 = 0; r6 < 6; ++r6) rr[r6] = *(const v8h*)(tb + 16 * r6);

        const size_t base = (size_t)img * (IMG_H * IMG_W) + (size_t)(y0 + nn) * IMG_W
                          + (size_t)(x0 + 8 * hs);
        v4f xa = *(const v4f*)(x + base);
        v4f xb = *(const v4f*)(x + base + 4);

        v8f acc = {};
#pragma unroll
        for (int kc = 0; kc < 3; ++kc) {
            v16h a;
#pragma unroll
            for (int e = 0; e < 8; ++e) {
                a[e]     = rr[kc * 2][e];
                a[8 + e] = rr[kc * 2 + 1][e];
            }
            acc = __builtin_amdgcn_wmma_f32_16x16x32_f16(false, a, false, bm[kc],
                                                         (short)0, acc, false, false);
        }

        // D[m][n] = out[y0+n][x0+m]: 8 contiguous fp32 per lane in row y0+nn.
        v4f o0, o1;
#pragma unroll
        for (int r = 0; r < 4; ++r) {
            float va = 1.0f - (1.0f - xa[r]) * (1.0f - acc[r] * s);
            float vb = 1.0f - (1.0f - xb[r]) * (1.0f - acc[4 + r] * s);
            o0[r] = fminf(fmaxf(va, 0.0f), 1.0f);
            o1[r] = fminf(fmaxf(vb, 0.0f), 1.0f);
        }
        // Non-temporal: don't let the 100MB output evict x/tmpT from L2.
        __builtin_nontemporal_store(o0, (v4f*)(out + base));
        __builtin_nontemporal_store(o1, (v4f*)(out + base + 4));
    }
}

extern "C" void kernel_launch(void* const* d_in, const int* in_sizes, int n_in,
                              void* d_out, int out_size, void* d_ws, size_t ws_size,
                              hipStream_t stream) {
    (void)in_sizes; (void)n_in; (void)out_size; (void)ws_size;
    const float* x        = (const float*)d_in[0];
    const float* amount_p = (const float*)d_in[1];
    float*       out      = (float*)d_out;
    _Float16*    tmpT     = (_Float16*)d_ws;   // needs NCHAN*IMG_W*HP*2 = ~51.8 MB

    {
        const int total = NCHAN * IMG_W * 80;
        orton_zero_pads<<<(total + 255) / 256, 256, 0, stream>>>(tmpT);
    }
    dim3 blk(256);
    dim3 grd(NWAVES / 8);   // 8 wave32 per block, TPW tiles per wave
    orton_pass1<<<grd, blk, 0, stream>>>(x, tmpT);
    orton_pass2<<<grd, blk, 0, stream>>>(x, tmpT, amount_p, out);
}
